// Conv1dFix_89644557402207
// MI455X (gfx1250) — matrix operand order repaired and checked by reading, not compiled, not run on hardware
//
#include <hip/hip_runtime.h>

typedef _Float16 v8h  __attribute__((ext_vector_type(8)));
typedef _Float16 v16h __attribute__((ext_vector_type(16)));
typedef float    v8f  __attribute__((ext_vector_type(8)));
typedef float    v4f  __attribute__((ext_vector_type(4)));

#define B_   16
#define C_   512
#define O_   512
#define T_   4096
#define KT   5                   // taps
#define TM   64                  // O tile per workgroup
#define TN   256                 // T tile per workgroup
#define TK   32                  // channel chunk
#define NCC  (C_ / TK)           // 16 chunks
#define XROWS (TN + 8)           // 264 rows: gt = t0-4 .. t0+259 (4-aligned halo)
#define XSTR  80                 // X row stride (bytes): conflict-free b128 frag loads
#define ASTR  80                 // A row stride (bytes)
#define ATAP  (TM * ASTR)        // 5120 bytes per tap
#define SX_BYTES (XROWS * XSTR)  // 21120
#define SA_BYTES (KT * ATAP)     // 25600  -> total 46720 B LDS
#define XGROUPS (XROWS / 4)      // 66 float4 groups per channel
#define XELEMS  (TK * XGROUPS)   // 2112 float4 loads per chunk
#define XV_N 9                   // ceil(2112/256) held in regs
#define WV_N 5                   // 1280/256 v8h (16B) weight vectors per thread

// ---------- prep: w[o][c][n] f32 -> wf[n][o][c] f16 (one-time, L2-resident) ----------
__global__ void wcvt_kernel(const float* __restrict__ w, _Float16* __restrict__ wf) {
  int i = blockIdx.x * 256 + threadIdx.x;
  if (i >= O_ * C_ * KT) return;
  int o = i / (C_ * KT);
  int r = i - o * (C_ * KT);
  int c = r / KT;
  int n = r - c * KT;
  wf[((size_t)n * O_ + o) * C_ + c] = (_Float16)w[i];
}

// Assemble a 16x32 f16 WMMA operand from two ds_load_b128 (per ISA VGPR layout).
__device__ __forceinline__ v16h frag_ld(const unsigned char* p, int half_off) {
  v8h lo = *(const v8h*)(p);
  v8h hi = *(const v8h*)(p + half_off);
  return __builtin_shufflevector(lo, hi, 0,1,2,3,4,5,6,7,8,9,10,11,12,13,14,15);
}

// X prefetch, k-range [K0,K1); GUARD=false for interior T-tiles (no bounds checks).
template <int K0, int K1, bool GUARD>
__device__ __forceinline__ void load_x_part(const float* __restrict__ x, int b, int ccb,
                                            int t0, int tid, v4f (&xv)[XV_N]) {
  #pragma unroll
  for (int k = K0; k < K1; ++k) {
    int e = tid + k * 256;
    v4f v = {0.0f, 0.0f, 0.0f, 0.0f};
    if (e < XELEMS) {
      int c  = e / XGROUPS;
      int g  = e - c * XGROUPS;          // consecutive tid -> consecutive t: coalesced
      int gt = t0 - 4 + g * 4;           // 16B aligned
      const float* p = x + (size_t)(b * C_ + ccb + c) * T_ + gt;
      if (!GUARD || (gt >= 0 && gt + 3 < T_)) {
        v = *(const v4f*)p;
      } else {
        #pragma unroll
        for (int j = 0; j < 4; ++j)
          if (gt + j >= 0 && gt + j < T_) v[j] = p[j];
      }
    }
    xv[k] = v;
  }
}

__device__ __forceinline__ void store_x(unsigned char* sx, int tid, const v4f (&xv)[XV_N]) {
  #pragma unroll
  for (int k = 0; k < XV_N; ++k) {
    int e = tid + k * 256;
    if (e < XELEMS) {
      int c = e / XGROUPS;
      int g = e - c * XGROUPS;
      unsigned char* p = sx + (g * 4) * XSTR + c * 2;
      #pragma unroll
      for (int j = 0; j < 4; ++j)
        *(_Float16*)(p + j * XSTR) = (_Float16)xv[k][j];
    }
  }
}

__device__ __forceinline__ void load_w(const _Float16* __restrict__ wf, int o0, int ccb,
                                       int tid, v8h (&wv)[WV_N]) {
  #pragma unroll
  for (int k = 0; k < WV_N; ++k) {
    int g  = tid + k * 256;              // 0..1279, exact
    int n  = g / (TM * 4);
    int r  = g - n * (TM * 4);
    int o  = r >> 2;
    int c8 = r & 3;
    wv[k] = *(const v8h*)(wf + ((size_t)n * O_ + o0 + o) * C_ + ccb + c8 * 8);
  }
}

__device__ __forceinline__ void store_w(unsigned char* sa, int tid, const v8h (&wv)[WV_N]) {
  #pragma unroll
  for (int k = 0; k < WV_N; ++k) {
    int g  = tid + k * 256;
    int n  = g / (TM * 4);
    int r  = g - n * (TM * 4);
    int o  = r >> 2;
    int c8 = r & 3;
    *(v8h*)(sa + n * ATAP + o * ASTR + c8 * 16) = wv[k];  // ds_store_b128
  }
}

// One tap: 2 A-frags x 4 B-frags -> 8 WMMAs (n is unrolled-constant at call sites).
__device__ __forceinline__ void tap_step(int n, const unsigned char* sa, const unsigned char* sx,
                                         int mB, int nB, int lr, int hs, v8f (&acc)[2][4]) {
  const unsigned char* ap = sa + n * ATAP + hs * 16;
  v16h a0 = frag_ld(ap + (mB +      lr) * ASTR, 32);
  v16h a1 = frag_ld(ap + (mB + 16 + lr) * ASTR, 32);
  const unsigned char* bp = sx + (nB + lr + 6 - n) * XSTR + hs * 16;
  #pragma unroll
  for (int fn = 0; fn < 4; ++fn) {
    v16h bf = frag_ld(bp + fn * 16 * XSTR, 32);
    acc[0][fn] = __builtin_amdgcn_wmma_f32_16x16x32_f16(false, a0, false, bf,
                                                        (short)0, acc[0][fn], false, false);
    acc[1][fn] = __builtin_amdgcn_wmma_f32_16x16x32_f16(false, a1, false, bf,
                                                        (short)0, acc[1][fn], false, false);
  }
}

__global__ __launch_bounds__(256)
void conv1d_wmma_f16(const float* __restrict__ x, const _Float16* __restrict__ wf,
                     const float* __restrict__ bias, float* __restrict__ out) {
  __shared__ __align__(16) unsigned char smem[SX_BYTES + SA_BYTES];
  unsigned char* sx = smem;              // X tile: [264 t][32 c] f16, stride 80B
  unsigned char* sa = smem + SX_BYTES;   // W tile: per tap [64 o][32 c] f16, stride 80B

  const int tid  = threadIdx.x;
  const int lane = tid & 31;             // wave32
  const int lr   = lane & 15;
  const int hs   = lane >> 4;
  const int wave = tid >> 5;
  const int mB   = (wave >> 2) * 32;     // 2 waves over O (32 rows each)
  const int nB   = (wave & 3)  * 64;     // 4 waves over T (64 cols each)

  const int t0 = blockIdx.x * TN;
  const int o0 = blockIdx.y * TM;
  const int b  = blockIdx.z;
  const bool edge = (blockIdx.x == 0) || (blockIdx.x == gridDim.x - 1);

  // preload the 16 bias values this lane needs (overlaps with staging)
  float breg[2][8];
  #pragma unroll
  for (int fm = 0; fm < 2; ++fm)
    #pragma unroll
    for (int r = 0; r < 8; ++r)
      breg[fm][r] = bias[o0 + mB + fm * 16 + hs * 8 + r];

  v8f acc[2][4] = {};                    // 32x64 per wave: 8 accumulators
  v4f xv[XV_N];
  v8h wv[WV_N];

  // prologue: stage chunk 0
  if (edge) load_x_part<0, XV_N, true >(x, b, 0, t0, tid, xv);
  else      load_x_part<0, XV_N, false>(x, b, 0, t0, tid, xv);
  load_w(wf, o0, 0, tid, wv);
  store_x(sx, tid, xv);
  store_w(sa, tid, wv);
  __syncthreads();

  #pragma unroll 1
  for (int cc = 0; cc < NCC; ++cc) {
    const int ccb = (cc + 1) * TK;
    const bool pf = (cc + 1 < NCC);

    // first half of next-chunk X prefetch, then 2 taps of WMMA to bury load waits
    if (pf) {
      if (edge) load_x_part<0, 5, true >(x, b, ccb, t0, tid, xv);
      else      load_x_part<0, 5, false>(x, b, ccb, t0, tid, xv);
    }
    #pragma unroll
    for (int n = 0; n < 2; ++n) tap_step(n, sa, sx, mB, nB, lr, hs, acc);

    // second half of X prefetch + W prefetch, then remaining 3 taps
    if (pf) {
      if (edge) load_x_part<5, XV_N, true >(x, b, ccb, t0, tid, xv);
      else      load_x_part<5, XV_N, false>(x, b, ccb, t0, tid, xv);
      load_w(wf, o0, ccb, tid, wv);
    }
    #pragma unroll
    for (int n = 2; n < KT; ++n) tap_step(n, sa, sx, mB, nB, lr, hs, acc);

    if (pf) {
      __syncthreads();                   // all waves done reading current LDS tile
      store_x(sx, tid, xv);
      store_w(sa, tid, wv);
      __syncthreads();                   // next tile visible to all waves
    }
  }

  // epilogue: C/D layout (VGPR r -> M = r + 8*hs, lane%16 -> N); add bias, store f32
  #pragma unroll
  for (int fm = 0; fm < 2; ++fm) {
    #pragma unroll
    for (int fn = 0; fn < 4; ++fn) {
      const int ob = o0 + mB + fm * 16 + hs * 8;
      const int t  = t0 + nB + fn * 16 + lr;
      #pragma unroll
      for (int r = 0; r < 8; ++r)
        out[((size_t)(b * O_) + ob + r) * T_ + t] = acc[fm][fn][r] + breg[fm][r];
    }
  }
}

extern "C" void kernel_launch(void* const* d_in, const int* in_sizes, int n_in,
                              void* d_out, int out_size, void* d_ws, size_t ws_size,
                              hipStream_t stream) {
  const float* x    = (const float*)d_in[0];   // [16,512,4096]
  const float* w    = (const float*)d_in[1];   // [512,512,5]
  const float* bias = (const float*)d_in[2];   // [512]
  float* out        = (float*)d_out;           // [16,512,4096]
  _Float16* wf      = (_Float16*)d_ws;         // [5][512][512] f16, 2.62 MB scratch

  // one-time weight convert/transpose (same stream -> ordered before main kernel)
  const int welems = O_ * C_ * KT;
  wcvt_kernel<<<(welems + 255) / 256, 256, 0, stream>>>(w, wf);

  dim3 grid(T_ / TN, O_ / TM, B_);             // 16 x 8 x 16 = 2048 workgroups
  conv1d_wmma_f16<<<grid, 256, 0, stream>>>(x, wf, bias, out);
}
